// FutureTrajDecoder_2362232013354
// MI455X (gfx1250) — matrix operand order repaired and checked by reading, not compiled, not run on hardware
//
#include <hip/hip_runtime.h>

// ---------------------------------------------------------------------------
// GRU decoder rollout for MI455X (gfx1250).
// B=16384 rows, ES=HS=64, 3 layers, T=128 steps, NOUT=2.
// One wave owns 16 rows for the whole rollout. All weights staged to LDS as
// bf16 with rows padded to 72 elements (144 B = 36 dwords; gcd(36,64)=4 ->
// conflict-free ds_load_b128 fragment fetches). GEMMs via
// v_wmma_f32_16x16x32_bf16 (K=64 -> 2 WMMAs per 16x16 tile), inline-0 C,
// biases folded into the gate nonlinearity as pre-hoisted registers,
// hardware v_tanh_f32 when available. B fragments for each tile are loaded
// as a 12-wide block ahead of the 12 WMMAs to enable partial dscnt waits.
// ---------------------------------------------------------------------------

typedef __bf16 bf16;
typedef __attribute__((ext_vector_type(8)))  bf16  bf16x8;
typedef __attribute__((ext_vector_type(16))) bf16  v16bf;
typedef __attribute__((ext_vector_type(8)))  float v8f;

namespace {
constexpr int kBatch = 16384;
constexpr int kES    = 64;
constexpr int kHS    = 64;
constexpr int kNL    = 3;
constexpr int kNOUT  = 2;
constexpr int kT     = 128;

constexpr int kWavesPerWG  = 8;
constexpr int kRowsPerWave = 16;
constexpr int kRowsPerWG   = kWavesPerWG * kRowsPerWave;   // 128
constexpr int kGrid        = kBatch / kRowsPerWG;          // 128

// Padded row stride (elements): 72 bf16 = 144 B = 36 dwords -> bank-conflict
// free b128 fragment loads (36 and 64 share only factor 4).
constexpr int kPad  = 72;
constexpr int kTile = kRowsPerWave * kPad;                 // 1152 bf16 per 16x64 tile

// Weight rows: emb 64, w_ih 3*192, w_hh 3*192, out 2
constexpr int kWembRows = kES;            // 64
constexpr int kWihRows  = kNL * 3 * kHS;  // 576
constexpr int kWhhRows  = kNL * 3 * kHS;  // 576
constexpr int kWoutRows = kNOUT;          // 2
constexpr int kWRows    = kWembRows + kWihRows + kWhhRows + kWoutRows; // 1218
constexpr int kWElems   = kWRows * kPad;  // 87696 bf16 = 175392 B

constexpr int kBiasN = kES + kNL * 3 * kHS + kNL * 3 * kHS + 8;  // f32, padded
constexpr int kWaveBufN = 5 * kTile;      // ping + pong + 3*h = 5760 bf16
constexpr size_t kLdsBytes =
    (size_t)kWElems * 2 + (size_t)kBiasN * 4 +
    (size_t)kWavesPerWG * kWaveBufN * 2;  // 175392 + 4896 + 92160 = 272448 B
} // namespace

__device__ __forceinline__ void lds_fence() {
  asm volatile("s_wait_dscnt 0" ::: "memory");
}

// Hardware tanh (TRANS op on CDNA5); compiler-managed hazards. Fallback: exp.
#if __has_builtin(__builtin_amdgcn_tanhf)
__device__ __forceinline__ float htanh(float x) { return __builtin_amdgcn_tanhf(x); }
#else
__device__ __forceinline__ float htanh(float x) {
  return 1.0f - 2.0f / (__expf(2.0f * x) + 1.0f);
}
#endif

__device__ __forceinline__ v16bf join16(bf16x8 a, bf16x8 b) {
  return __builtin_shufflevector(a, b, 0, 1, 2, 3, 4, 5, 6, 7,
                                 8, 9, 10, 11, 12, 13, 14, 15);
}

// A fragment (16x32 bf16) from a row-major [16][kPad] bf16 LDS tile, K-block kb.
// lanes 0-15 : row = lane,    K = kb*32 + {0..7, 16..23}
// lanes 16-31: row = lane-16, K = kb*32 + {8..15, 24..31}
__device__ __forceinline__ v16bf ldA(const bf16* buf, int kb, int lane) {
  const int hi  = (lane >> 4) & 1;
  const int row = lane & 15;
  const int e0  = row * kPad + kb * 32 + hi * 8;
  return join16(*(const bf16x8*)(buf + e0), *(const bf16x8*)(buf + e0 + 16));
}

// B fragment (32x16 bf16) from row-major [N][kPad] bf16 weights (W[n][k]),
// column block at nbase. lane holds col = nbase + (lane&15),
// K = kb*32 + (lane>=16 ? {8..15, 24..31} : {0..7, 16..23}).
__device__ __forceinline__ v16bf ldBf(const bf16* w, int nbase, int kb, int lane) {
  const int hi = (lane >> 4) & 1;
  const int n  = nbase + (lane & 15);
  const int e0 = n * kPad + kb * 32 + hi * 8;
  return join16(*(const bf16x8*)(w + e0), *(const bf16x8*)(w + e0 + 16));
}

__device__ __forceinline__ v8f wmma_bf16(v16bf a, v16bf b, v8f c) {
  return __builtin_amdgcn_wmma_f32_16x16x32_bf16(
      /*neg_a=*/false, a, /*neg_b=*/false, b,
      /*c_mod=*/(short)0, c, /*reuse_a=*/false, /*reuse_b=*/false);
}

__global__ __launch_bounds__(kWavesPerWG * 32, 1)
void gru_decoder_kernel(const float* __restrict__ enc,
                        const float* __restrict__ emb_w,
                        const float* __restrict__ emb_b,
                        const float* __restrict__ w_ih,
                        const float* __restrict__ w_hh,
                        const float* __restrict__ b_ih,
                        const float* __restrict__ b_hh,
                        const float* __restrict__ out_w,
                        const float* __restrict__ out_b,
                        float* __restrict__ out) {
  extern __shared__ char smem[];
  bf16* Wemb = (bf16*)smem;                       // 64 rows
  bf16* Wih  = Wemb + kWembRows * kPad;           // 576 rows
  bf16* Whh  = Wih + kWihRows * kPad;             // 576 rows
  bf16* Wout = Whh + kWhhRows * kPad;             // 2 rows
  float* Bemb = (float*)(smem + (size_t)kWElems * 2);
  float* Bih  = Bemb + kES;
  float* Bhh  = Bih + kNL * 3 * kHS;
  float* Bout = Bhh + kNL * 3 * kHS;
  bf16* wbufs = (bf16*)(smem + (size_t)kWElems * 2 + (size_t)kBiasN * 4);

  const int tid  = threadIdx.x;
  const int lane = tid & 31;
  const int wave = tid >> 5;
  const int hi   = (lane >> 4) & 1;
  const int col  = lane & 15;

  // ---- cooperative weight staging (f32 -> bf16, padded rows) ---------------
  for (int i = tid; i < kWembRows * kES; i += kWavesPerWG * 32)
    Wemb[(i >> 6) * kPad + (i & 63)] = (bf16)emb_w[i];
  for (int i = tid; i < kWihRows * kES; i += kWavesPerWG * 32) {
    const int d = (i >> 6) * kPad + (i & 63);
    Wih[d] = (bf16)w_ih[i];
    Whh[d] = (bf16)w_hh[i];
  }
  for (int i = tid; i < kWoutRows * kES; i += kWavesPerWG * 32)
    Wout[(i >> 6) * kPad + (i & 63)] = (bf16)out_w[i];
  for (int i = tid; i < kES; i += kWavesPerWG * 32) Bemb[i] = emb_b[i];
  for (int i = tid; i < kNL * 3 * kHS; i += kWavesPerWG * 32) {
    Bih[i] = b_ih[i];
    Bhh[i] = b_hh[i];
  }
  if (tid < kNOUT) Bout[tid] = out_b[tid];
  __syncthreads();

  // ---- per-lane bias registers, hoisted ahead of the t-loop ----------------
  // sigmoid(x) = 0.5 + 0.5*tanh(0.5*x)  -> pre-scale r/z biases by 0.5.
  float be[4];
  float brz[kNL][4], bzz[kNL][4], bin[kNL][4], bhn[kNL][4];
#pragma unroll
  for (int nt = 0; nt < 4; ++nt) be[nt] = Bemb[nt * 16 + col];
#pragma unroll
  for (int l = 0; l < kNL; ++l)
#pragma unroll
    for (int nt = 0; nt < 4; ++nt) {
      const int o = l * 192 + nt * 16 + col;
      brz[l][nt] = 0.5f * (Bih[o] + Bhh[o]);
      bzz[l][nt] = 0.5f * (Bih[o + 64] + Bhh[o + 64]);
      bin[l][nt] = Bih[o + 128];
      bhn[l][nt] = Bhh[o + 128];
    }
  const float outb = Bout[lane & 1];

  // ---- per-wave buffers ----
  bf16* mybuf = wbufs + wave * kWaveBufN;
  bf16* bufA  = mybuf;              // x / inp ping
  bf16* bufB  = mybuf + kTile;      // pong
  bf16* hbuf  = mybuf + 2 * kTile;  // h[l] = hbuf + l*kTile

  const int row0 = blockIdx.x * kRowsPerWG + wave * kRowsPerWave;

  // init: x = enc tile; h[l] = enc tile (broadcast across layers)
  for (int i = lane; i < 16 * 64; i += 32) {
    const bf16 bv = (bf16)enc[(size_t)row0 * kES + i];
    const int  d  = (i >> 6) * kPad + (i & 63);
    bufA[d] = bv;
    hbuf[d] = bv;
    hbuf[kTile + d] = bv;
    hbuf[2 * kTile + d] = bv;
  }
  lds_fence();

  bf16* cur = bufA;
  bf16* nxt = bufB;

#pragma unroll 1
  for (int t = 0; t < kT; ++t) {
    // ================= embedding: inp = x @ emb_w^T + emb_b =================
    {
      const v16bf xA0 = ldA(cur, 0, lane);
      const v16bf xA1 = ldA(cur, 1, lane);
      v16bf eb[8];
#pragma unroll
      for (int nt = 0; nt < 4; ++nt) {
        eb[2 * nt]     = ldBf(Wemb, nt * 16, 0, lane);
        eb[2 * nt + 1] = ldBf(Wemb, nt * 16, 1, lane);
      }
#pragma unroll
      for (int nt = 0; nt < 4; ++nt) {
        v8f acc = {};  // inline-0 C operand
        acc = wmma_bf16(xA0, eb[2 * nt], acc);
        acc = wmma_bf16(xA1, eb[2 * nt + 1], acc);
#pragma unroll
        for (int rr = 0; rr < 8; ++rr)
          nxt[(rr + 8 * hi) * kPad + nt * 16 + col] = (bf16)(acc[rr] + be[nt]);
      }
      lds_fence();
    }

    // ================= GRU layers =================
#pragma unroll
    for (int l = 0; l < kNL; ++l) {
      const bf16* Wi = Wih + l * (3 * kHS) * kPad;
      const bf16* Wh = Whh + l * (3 * kHS) * kPad;
      bf16* hb = hbuf + l * kTile;

      const v16bf xA0 = ldA(nxt, 0, lane);
      const v16bf xA1 = ldA(nxt, 1, lane);
      const v16bf hA0 = ldA(hb, 0, lane);
      const v16bf hA1 = ldA(hb, 1, lane);

#pragma unroll
      for (int nt = 0; nt < 4; ++nt) {
        const int nb = nt * 16;

        // Block-load all 12 B fragments for this tile (deep dscnt window).
        v16bf B[12];
        B[0]  = ldBf(Wi, nb, 0, lane);        // r: gi
        B[1]  = ldBf(Wi, nb, 1, lane);
        B[2]  = ldBf(Wh, nb, 0, lane);        // r: gh
        B[3]  = ldBf(Wh, nb, 1, lane);
        B[4]  = ldBf(Wi, 64 + nb, 0, lane);   // z: gi
        B[5]  = ldBf(Wi, 64 + nb, 1, lane);
        B[6]  = ldBf(Wh, 64 + nb, 0, lane);   // z: gh
        B[7]  = ldBf(Wh, 64 + nb, 1, lane);
        B[8]  = ldBf(Wi, 128 + nb, 0, lane);  // n: inn
        B[9]  = ldBf(Wi, 128 + nb, 1, lane);
        B[10] = ldBf(Wh, 128 + nb, 0, lane);  // n: hn
        B[11] = ldBf(Wh, 128 + nb, 1, lane);

        // Four independent accumulation chains, C starts at inline 0.
        v8f r = {};
        r = wmma_bf16(xA0, B[0], r);
        r = wmma_bf16(xA1, B[1], r);
        r = wmma_bf16(hA0, B[2], r);
        r = wmma_bf16(hA1, B[3], r);

        v8f z = {};
        z = wmma_bf16(xA0, B[4], z);
        z = wmma_bf16(xA1, B[5], z);
        z = wmma_bf16(hA0, B[6], z);
        z = wmma_bf16(hA1, B[7], z);

        v8f inn = {};
        inn = wmma_bf16(xA0, B[8], inn);
        inn = wmma_bf16(xA1, B[9], inn);

        v8f hn = {};
        hn = wmma_bf16(hA0, B[10], hn);
        hn = wmma_bf16(hA1, B[11], hn);

        // Elementwise gates + immediate commit. Safe: A operands of inp/h are
        // already in registers, and this tile writes only columns nb..nb+15,
        // which no later tile of this layer reads.
#pragma unroll
        for (int rr = 0; rr < 8; ++rr) {
          const float rg = __builtin_fmaf(
              0.5f, htanh(__builtin_fmaf(0.5f, r[rr], brz[l][nt])), 0.5f);
          const float zg = __builtin_fmaf(
              0.5f, htanh(__builtin_fmaf(0.5f, z[rr], bzz[l][nt])), 0.5f);
          const float ng = htanh(__builtin_fmaf(
              rg, hn[rr] + bhn[l][nt], inn[rr] + bin[l][nt]));
          const int  idx = (rr + 8 * hi) * kPad + nb + col;
          const float hv = (float)hb[idx];
          // (1-z)*n + z*h  ==  fma(z, h-n, n)
          const bf16 bv = (bf16)__builtin_fmaf(zg, hv - ng, ng);
          hb[idx]  = bv;
          nxt[idx] = bv;
        }
      }
      lds_fence();
    }

    // ================= output head: out = h_top @ out_w^T + out_b ==========
    {
      const int m = lane >> 1;        // 16 rows
      const int n = lane & 1;         // 2 output columns
      float acc = outb;
      const bf16* ip = nxt + m * kPad;
      const bf16* wr = Wout + n * kPad;
#pragma unroll
      for (int k8 = 0; k8 < 8; ++k8) {
        const bf16x8 a = *(const bf16x8*)(ip + k8 * 8);
        const bf16x8 b = *(const bf16x8*)(wr + k8 * 8);
#pragma unroll
        for (int j = 0; j < 8; ++j) acc = __builtin_fmaf((float)a[j], (float)b[j], acc);
      }
      out[((size_t)(row0 + m) * kT + t) * kNOUT + n] = acc;
    }

    // next x = top-layer hidden (currently in nxt)
    bf16* tmp = cur;
    cur = nxt;
    nxt = tmp;
  }
}

extern "C" void kernel_launch(void* const* d_in, const int* in_sizes, int n_in,
                              void* d_out, int out_size, void* d_ws, size_t ws_size,
                              hipStream_t stream) {
  (void)in_sizes; (void)n_in; (void)d_ws; (void)ws_size; (void)out_size;

  const float* enc   = (const float*)d_in[0];
  const float* emb_w = (const float*)d_in[1];
  const float* emb_b = (const float*)d_in[2];
  const float* w_ih  = (const float*)d_in[3];
  const float* w_hh  = (const float*)d_in[4];
  const float* b_ih  = (const float*)d_in[5];
  const float* b_hh  = (const float*)d_in[6];
  const float* out_w = (const float*)d_in[7];
  const float* out_b = (const float*)d_in[8];
  float* out = (float*)d_out;

  // 272448 B dynamic LDS (> 64 KB default cap; WGP has 320 KB). Deterministic
  // per-call, not a stream op, safe under graph capture.
  (void)hipFuncSetAttribute((const void*)gru_decoder_kernel,
                            hipFuncAttributeMaxDynamicSharedMemorySize,
                            (int)kLdsBytes);

  gru_decoder_kernel<<<kGrid, kWavesPerWG * 32, kLdsBytes, stream>>>(
      enc, emb_w, emb_b, w_ih, w_hh, b_ih, b_hh, out_w, out_b, out);
}
